// RecurrentSelfAttentionLayer_17171279249477
// MI455X (gfx1250) — matrix operand order repaired and checked by reading, not compiled, not run on hardware
//
#include <hip/hip_runtime.h>
#include <hip/hip_bf16.h>

// Problem constants (from reference)
#define B_   4
#define L_   2048
#define DM_  1024
#define H_   16
#define HD_  64
#define NC_  32              // L / 64 chunks
#define BH_  (B_ * H_)       // 64
#define M_   (B_ * L_)       // 8192
#define EPS_ 1e-6f

typedef __attribute__((ext_vector_type(16))) __bf16 v16bf;
typedef __attribute__((ext_vector_type(8)))  __bf16 v8bf;
typedef __attribute__((ext_vector_type(4)))  __bf16 v4bf;
typedef __attribute__((ext_vector_type(4)))  float  v4f;
typedef __attribute__((ext_vector_type(8)))  float  v8f;

// Native bf16 converts (single v_cvt ops on gfx1250)
__device__ __forceinline__ __bf16 f2bf(float f) { return (__bf16)f; }
__device__ __forceinline__ float  bf2f(__bf16 b) { return (float)b; }

__device__ __forceinline__ v8f wmma_bf16(v16bf a, v16bf b, v8f c) {
    return __builtin_amdgcn_wmma_f32_16x16x32_bf16(
        false, a, false, b, (short)0, c, false, false);
}

// ---------------------------------------------------------------------------
// Wave32 WMMA fragment loaders, vectorized (2x ds_load_b128 each).
__device__ __forceinline__ v16bf load_a_frag(const __bf16* lds, int ld, int m0, int k0) {
    const int lane = threadIdx.x & 31;
    const __bf16* p = lds + (m0 + (lane & 15)) * ld + k0 + ((lane >> 4) << 3);
    v8bf lo = *(const v8bf*)p;
    v8bf hi = *(const v8bf*)(p + 16);
    return __builtin_shufflevector(lo, hi, 0,1,2,3,4,5,6,7,8,9,10,11,12,13,14,15);
}
__device__ __forceinline__ v16bf load_b_frag(const __bf16* lds, int ld, int k0, int n0) {
    const int lane = threadIdx.x & 31;
    const __bf16* p = lds + (n0 + (lane & 15)) * ld + k0 + ((lane >> 4) << 3);
    v8bf lo = *(const v8bf*)p;
    v8bf hi = *(const v8bf*)(p + 16);
    return __builtin_shufflevector(lo, hi, 0,1,2,3,4,5,6,7,8,9,10,11,12,13,14,15);
}

// ===========================================================================
// Kernel 1/5: Y = X @ W + bias.  BM=128, BN=64, BK=32, double-buffered LDS.
//   MODE 0: q-proj  -> (y*scale, elu+1) -> bf16, (B,H,L,HD)
//   MODE 1: k-proj  -> (elu+1)          -> bf16, (B,H,L,HD)
//   MODE 2: v-proj                      -> bf16, (B,H,L,HD)
//   MODE 3: o-proj  (X is bf16)         -> f32, row-major (B,L,DM)
// Modes 0-2 stage the output tile in LDS for coalesced b128 stores
// (BN == HD, so each block's n-range is exactly one head).
// ===========================================================================
template <int MODE>
__global__ __launch_bounds__(256)
void gemm_proj(const float* __restrict__ X, const __bf16* __restrict__ Xb,
               const float* __restrict__ W, const float* __restrict__ bias,
               float scaling,
               __bf16* __restrict__ outB, float* __restrict__ outF)
{
    __shared__ __align__(16) __bf16 Xs[2][128][40];  // [m][k], k-contiguous
    __shared__ __align__(16) __bf16 Wt[2][64][40];   // [n][k], transposed
    __shared__ __align__(16) __bf16 Osh[128][72];    // output staging (modes 0-2)
    const int tid = threadIdx.x;
    const int m0 = blockIdx.x * 128;
    const int n0 = blockIdx.y * 64;
    const int wid = tid >> 5;                        // wave owns m-strip wid*16

    v8f acc[4] = {{}, {}, {}, {}};

    auto load_tile = [&](int kk, int buf) {
        if (MODE == 3) {
#pragma unroll
            for (int i = tid; i < 512; i += 256) {          // 128 rows x 4 v8bf
                int r = i >> 2, c8 = (i & 3) << 3;
                *(v8bf*)&Xs[buf][r][c8] = *(const v8bf*)&Xb[(size_t)(m0 + r) * DM_ + kk + c8];
            }
        } else {
#pragma unroll
            for (int i = tid; i < 1024; i += 256) {         // 128 rows x 8 float4
                int r = i >> 3, c4 = (i & 7) << 2;
                float4 x = *(const float4*)&X[(size_t)(m0 + r) * DM_ + kk + c4];
                v4f xf = {x.x, x.y, x.z, x.w};
                *(v4bf*)&Xs[buf][r][c4] = __builtin_convertvector(xf, v4bf);
            }
        }
#pragma unroll
        for (int i = tid; i < 512; i += 256) {              // 32 k-rows x 16 float4
            int r = i >> 4, c4 = (i & 15) << 2;
            float4 w4 = *(const float4*)&W[(size_t)(kk + r) * DM_ + n0 + c4];
            Wt[buf][c4 + 0][r] = f2bf(w4.x);
            Wt[buf][c4 + 1][r] = f2bf(w4.y);
            Wt[buf][c4 + 2][r] = f2bf(w4.z);
            Wt[buf][c4 + 3][r] = f2bf(w4.w);
        }
    };

    load_tile(0, 0);
    int cur = 0;
    for (int kk = 0; kk < DM_; kk += 32) {
        __syncthreads();                                    // buf[cur] ready / safe
        if (kk + 32 < DM_) load_tile(kk + 32, cur ^ 1);
        if (kk + 64 < DM_) {                                // prefetch tile after next
            __builtin_prefetch(&W[(size_t)(kk + 64 + (tid & 31)) * DM_ + n0], 0, 1);
            if (MODE == 3) __builtin_prefetch(&Xb[(size_t)(m0 + (tid >> 1)) * DM_ + kk + 64], 0, 1);
            else           __builtin_prefetch(&X [(size_t)(m0 + (tid >> 1)) * DM_ + kk + 64], 0, 1);
        }
        v16bf a = load_a_frag(&Xs[cur][0][0], 40, wid * 16, 0);
#pragma unroll
        for (int j = 0; j < 4; ++j) {
            v16bf b = load_b_frag(&Wt[cur][0][0], 40, 0, j * 16);
            acc[j] = wmma_bf16(a, b, acc[j]);
        }
        cur ^= 1;
    }

    const int lane  = tid & 31;
    const int nloc  = lane & 15;
    const int mbase = (lane >> 4) << 3;
#pragma unroll
    for (int j = 0; j < 4; ++j) {
        int n  = j * 16 + nloc;
        int gn = n0 + n;
        float bb = bias[gn];
#pragma unroll
        for (int r = 0; r < 8; ++r) {
            int m  = wid * 16 + mbase + r;
            int gm = m0 + m;
            float y = acc[j][r] + bb;
            if (MODE == 0) { y *= scaling; y = (y > 0.f) ? y + 1.f : __expf(y); }
            if (MODE == 1) { y = (y > 0.f) ? y + 1.f : __expf(y); }
            if (MODE == 3) {
                outF[(size_t)gm * DM_ + gn] = y;
            } else {
                Osh[m][n] = f2bf(y);
            }
        }
    }
    if (MODE != 3) {                                        // coalesced b128 stores
        __syncthreads();
        const int hh = n0 >> 6;                             // one head per block
#pragma unroll
        for (int i = tid; i < 1024; i += 256) {
            int r = i >> 3, d8 = (i & 7) << 3;
            int gm = m0 + r;
            int bidx = gm >> 11, t = gm & (L_ - 1);
            *(v8bf*)&outB[(((size_t)(bidx * H_ + hh)) * L_ + t) * HD_ + d8] =
                *(const v8bf*)&Osh[r][d8];
        }
    }
}

// ===========================================================================
// Kernel 2: per-chunk sums  S_c = K_chunk^T @ V_chunk,  Z_c = colsum(K)
// ===========================================================================
__global__ __launch_bounds__(256)
void chunk_sums(const __bf16* __restrict__ Kf, const __bf16* __restrict__ Vp,
                float* __restrict__ Sp, float* __restrict__ Zp)
{
    __shared__ __align__(16) __bf16 Kt[64][72];
    __shared__ __align__(16) __bf16 Vt[64][72];
    const int tid = threadIdx.x;
    const int bh = blockIdx.x >> 5;
    const int c  = blockIdx.x & 31;
    const size_t base = ((size_t)bh * L_ + c * 64) * HD_;
#pragma unroll
    for (int i = tid; i < 512; i += 256) {        // transpose in
        int t = i >> 3, d8 = (i & 7) << 3;
        v8bf kv = *(const v8bf*)&Kf[base + (size_t)t * HD_ + d8];
        v8bf vv = *(const v8bf*)&Vp[base + (size_t)t * HD_ + d8];
#pragma unroll
        for (int j = 0; j < 8; ++j) { Kt[d8 + j][t] = kv[j]; Vt[d8 + j][t] = vv[j]; }
    }
    __syncthreads();

    const int wid = tid >> 5, tm = wid & 3, tn0 = wid >> 2;
    v8f acc0 = {}, acc1 = {};
#pragma unroll
    for (int kt = 0; kt < 64; kt += 32) {
        v16bf a  = load_a_frag(&Kt[0][0], 72, tm * 16, kt);       // A(d,t)
        v16bf b0 = load_b_frag(&Vt[0][0], 72, kt, tn0 * 16);      // B(t,e)
        acc0 = wmma_bf16(a, b0, acc0);
        v16bf b1 = load_b_frag(&Vt[0][0], 72, kt, (tn0 + 2) * 16);
        acc1 = wmma_bf16(a, b1, acc1);
    }
    float* Sout = Sp + ((size_t)bh * NC_ + c) * (HD_ * HD_);
    const int lane = tid & 31, nloc = lane & 15, mbase = (lane >> 4) << 3;
#pragma unroll
    for (int j = 0; j < 2; ++j) {
        v8f acc = j ? acc1 : acc0;
        int n = (tn0 + 2 * j) * 16 + nloc;
#pragma unroll
        for (int r = 0; r < 8; ++r) Sout[(tm * 16 + mbase + r) * HD_ + n] = acc[r];
    }
    if (tid < 64) {                               // vectorized row reduction
        v8f sv = {};
#pragma unroll
        for (int j = 0; j < 8; ++j) {
            v8bf kv = *(const v8bf*)&Kt[tid][j * 8];
            sv += __builtin_convertvector(kv, v8f);
        }
        float s = 0.f;
#pragma unroll
        for (int e = 0; e < 8; ++e) s += sv[e];
        Zp[((size_t)bh * NC_ + c) * HD_ + tid] = s;
    }
}

// ===========================================================================
// Kernel 3: in-place exclusive prefix over the 32 chunk states per (b,h).
// ===========================================================================
__global__ __launch_bounds__(256)
void prefix_states(float* __restrict__ Sp, float* __restrict__ Zp)
{
    const int bh = blockIdx.x;
    const int tid = threadIdx.x;
    v4f run[4] = {{}, {}, {}, {}};
    float rz = 0.f;
    for (int c = 0; c < NC_; ++c) {
        v4f* Sc4 = (v4f*)(Sp + ((size_t)bh * NC_ + c) * (HD_ * HD_));
#pragma unroll
        for (int j = 0; j < 4; ++j) {
            int idx = tid + j * 256;             // 1024 float4s per state
            v4f cur = Sc4[idx];
            Sc4[idx] = run[j];
            run[j] += cur;
        }
        if (tid < HD_) {
            float* Zc = Zp + ((size_t)bh * NC_ + c) * HD_;
            float cur = Zc[tid];
            Zc[tid] = rz;
            rz += cur;
        }
    }
}

// ===========================================================================
// Kernel 4: per-chunk output
//   A = tril(Q K^T); den = q.Zpre + rowsum(A) + eps;  O = (Q Spre + A V)/den
// ===========================================================================
__global__ __launch_bounds__(256)
void chunk_out(const __bf16* __restrict__ Qf, const __bf16* __restrict__ Kf,
               const __bf16* __restrict__ Vp, const float* __restrict__ Sp,
               const float* __restrict__ Zp, __bf16* __restrict__ Ao)
{
    __shared__ __align__(16) __bf16 Qc[64][72], Kc[64][72], Vt[64][72], Sb[64][72], Ab[64][72];
    __shared__ __align__(16) float zsh[64];
    __shared__ float den[64];
    const int tid = threadIdx.x;
    const int bh = blockIdx.x >> 5;
    const int c  = blockIdx.x & 31;
    const int b  = bh >> 4, h = bh & 15;
    const int t0 = c * 64;
    const size_t base = ((size_t)bh * L_ + t0) * HD_;
    const float* Sc = Sp + ((size_t)bh * NC_ + c) * (HD_ * HD_);
#pragma unroll
    for (int i = tid; i < 512; i += 256) {
        int t = i >> 3, d8 = (i & 7) << 3;
        *(v8bf*)&Qc[t][d8] = *(const v8bf*)&Qf[base + (size_t)t * HD_ + d8];
        *(v8bf*)&Kc[t][d8] = *(const v8bf*)&Kf[base + (size_t)t * HD_ + d8];
        v8bf vv = *(const v8bf*)&Vp[base + (size_t)t * HD_ + d8];
#pragma unroll
        for (int j = 0; j < 8; ++j) Vt[d8 + j][t] = vv[j];
    }
#pragma unroll
    for (int i = tid; i < 1024; i += 256) {        // S (d,e) f32 -> Sb[e][d] bf16
        int d = i >> 4, e4 = (i & 15) << 2;
        float4 s4 = *(const float4*)&Sc[d * 64 + e4];
        Sb[e4 + 0][d] = f2bf(s4.x);
        Sb[e4 + 1][d] = f2bf(s4.y);
        Sb[e4 + 2][d] = f2bf(s4.z);
        Sb[e4 + 3][d] = f2bf(s4.w);
    }
    if (tid < 64) zsh[tid] = Zp[((size_t)bh * NC_ + c) * HD_ + tid];
    __syncthreads();

    const int wid = tid >> 5, tm = wid & 3, tn0 = wid >> 2;
    const int lane = tid & 31, nloc = lane & 15, mbase = (lane >> 4) << 3;

    // Stage A: masked scores into Ab (bf16); masked-out entries = exact 0
    {
        v8f a0 = {}, a1 = {};
#pragma unroll
        for (int kt = 0; kt < 64; kt += 32) {
            v16bf a  = load_a_frag(&Qc[0][0], 72, tm * 16, kt);
            v16bf b0 = load_b_frag(&Kc[0][0], 72, kt, tn0 * 16);
            a0 = wmma_bf16(a, b0, a0);
            v16bf b1 = load_b_frag(&Kc[0][0], 72, kt, (tn0 + 2) * 16);
            a1 = wmma_bf16(a, b1, a1);
        }
        __bf16 zb = f2bf(0.f);
#pragma unroll
        for (int j = 0; j < 2; ++j) {
            v8f acc = j ? a1 : a0;
            int n = (tn0 + 2 * j) * 16 + nloc;
#pragma unroll
            for (int r = 0; r < 8; ++r) {
                int m = tm * 16 + mbase + r;
                Ab[m][n] = (n <= m) ? f2bf(acc[r]) : zb;    // causal mask
            }
        }
    }
    __syncthreads();

    // denominators: q.Zpre + full-row sum of Ab (upper part is exact 0)
    if (tid < 64) {
        float s = EPS_;
#pragma unroll
        for (int j = 0; j < 16; ++j) {
            v4f qf = __builtin_convertvector(*(const v4bf*)&Qc[tid][j * 4], v4f);
            float4 z4 = *(const float4*)&zsh[j * 4];
            v4f af = __builtin_convertvector(*(const v4bf*)&Ab[tid][j * 4], v4f);
            s += qf[0] * z4.x + qf[1] * z4.y + qf[2] * z4.z + qf[3] * z4.w;
            s += af[0] + af[1] + af[2] + af[3];
        }
        den[tid] = s;
    }
    __syncthreads();

    // Stage C: O = Q @ Sprefix + Ab @ V, normalized
    v8f o0 = {}, o1 = {};
#pragma unroll
    for (int kt = 0; kt < 64; kt += 32) {
        v16bf aq  = load_a_frag(&Qc[0][0], 72, tm * 16, kt);
        v16bf bs0 = load_b_frag(&Sb[0][0], 72, kt, tn0 * 16);
        o0 = wmma_bf16(aq, bs0, o0);
        v16bf bs1 = load_b_frag(&Sb[0][0], 72, kt, (tn0 + 2) * 16);
        o1 = wmma_bf16(aq, bs1, o1);
        v16bf aa  = load_a_frag(&Ab[0][0], 72, tm * 16, kt);
        v16bf bv0 = load_b_frag(&Vt[0][0], 72, kt, tn0 * 16);
        o0 = wmma_bf16(aa, bv0, o0);
        v16bf bv1 = load_b_frag(&Vt[0][0], 72, kt, (tn0 + 2) * 16);
        o1 = wmma_bf16(aa, bv1, o1);
    }
    __syncthreads();                               // done reading Ab -> reuse as O staging
#pragma unroll
    for (int j = 0; j < 2; ++j) {
        v8f acc = j ? o1 : o0;
        int n = (tn0 + 2 * j) * 16 + nloc;
#pragma unroll
        for (int r = 0; r < 8; ++r) {
            int m = tm * 16 + mbase + r;
            Ab[m][n] = f2bf(acc[r] / den[m]);
        }
    }
    __syncthreads();
#pragma unroll
    for (int i = tid; i < 512; i += 256) {         // coalesced b128 stores
        int t = i >> 3, d8 = (i & 7) << 3;
        *(v8bf*)&Ao[((size_t)b * L_ + t0 + t) * DM_ + h * HD_ + d8] =
            *(const v8bf*)&Ab[t][d8];
    }
}

// ===========================================================================
extern "C" void kernel_launch(void* const* d_in, const int* in_sizes, int n_in,
                              void* d_out, int out_size, void* d_ws, size_t ws_size,
                              hipStream_t stream) {
    (void)in_sizes; (void)n_in; (void)out_size; (void)ws_size;
    const float* querys = (const float*)d_in[0];
    const float* keys   = (const float*)d_in[1];
    const float* values = (const float*)d_in[2];
    const float* Wq = (const float*)d_in[3];
    const float* bq = (const float*)d_in[4];
    const float* Wk = (const float*)d_in[5];
    const float* bk = (const float*)d_in[6];
    const float* Wv = (const float*)d_in[7];
    const float* bv = (const float*)d_in[8];
    const float* Wo = (const float*)d_in[9];
    const float* bo = (const float*)d_in[10];

    char* ws = (char*)d_ws;
    size_t off = 0;
    __bf16* Qf = (__bf16*)(ws + off); off += (size_t)M_ * DM_ * sizeof(__bf16);
    __bf16* Kf = (__bf16*)(ws + off); off += (size_t)M_ * DM_ * sizeof(__bf16);
    __bf16* Vp = (__bf16*)(ws + off); off += (size_t)M_ * DM_ * sizeof(__bf16);
    float*  Sp = (float*)(ws + off);  off += (size_t)BH_ * NC_ * HD_ * HD_ * sizeof(float);
    float*  Zp = (float*)(ws + off);  off += (size_t)BH_ * NC_ * HD_ * sizeof(float);
    __bf16* Ao = (__bf16*)(ws + off); off += (size_t)M_ * DM_ * sizeof(__bf16);

    dim3 blk(256);
    dim3 ggemm(M_ / 128, DM_ / 64);
    const float scaling = 0.125f;  // HD^-0.5

    gemm_proj<0><<<ggemm, blk, 0, stream>>>(querys, nullptr, Wq, bq, scaling, Qf, nullptr);
    gemm_proj<1><<<ggemm, blk, 0, stream>>>(keys,   nullptr, Wk, bk, 1.0f,    Kf, nullptr);
    gemm_proj<2><<<ggemm, blk, 0, stream>>>(values, nullptr, Wv, bv, 1.0f,    Vp, nullptr);
    chunk_sums   <<<dim3(BH_ * NC_), blk, 0, stream>>>(Kf, Vp, Sp, Zp);
    prefix_states<<<dim3(BH_),       blk, 0, stream>>>(Sp, Zp);
    chunk_out    <<<dim3(BH_ * NC_), blk, 0, stream>>>(Qf, Kf, Vp, Sp, Zp, Ao);
    gemm_proj<3><<<ggemm, blk, 0, stream>>>(nullptr, Ao, Wo, bo, 1.0f, nullptr, (float*)d_out);
}
